// UpDoubleConvertibleStyleBlock_34540126995022
// MI455X (gfx1250) — matrix operand order repaired and checked by reading, not compiled, not run on hardware
//
#include <hip/hip_runtime.h>
#include <hip/hip_bf16.h>

typedef __attribute__((ext_vector_type(16))) __bf16 v16bf;
typedef __attribute__((ext_vector_type(8)))  float  v8f;
typedef __attribute__((ext_vector_type(8)))  int    v8i;
typedef __attribute__((ext_vector_type(4)))  int    v4i;

static __device__ __forceinline__ unsigned short f2bf(float f) {
    unsigned int u = __float_as_uint(f);
    u += 0x7fffu + ((u >> 16) & 1u);
    return (unsigned short)(u >> 16);
}

static __device__ __forceinline__ float mishf(float x) {
    float sp = (x > 20.f) ? x : log1pf(expf(x));
    return x * tanhf(sp);
}

// ---------------------------------------------------------------------------
// Kernel 1: style vectors  s = z @ swT + sb + 1.0   -> (8,64) each
// ---------------------------------------------------------------------------
__global__ void style_y_kernel(const float* __restrict__ z,
                               const float* __restrict__ w1, const float* __restrict__ b1,
                               const float* __restrict__ w2, const float* __restrict__ b2,
                               float* __restrict__ s1, float* __restrict__ s2) {
    int b = blockIdx.x;          // 0..7
    int which = blockIdx.y;      // 0 -> style1, 1 -> style2
    int c = threadIdx.x;         // 0..63
    const float* w  = which ? w2 : w1;
    const float* bb = which ? b2 : b1;
    const float* zr = z + b * 512;
    const float* wr = w + c * 512;
    float acc = 0.f;
    for (int k = 0; k < 512; ++k) acc += zr[k] * wr[k];
    float v = acc + bb[c] + 1.0f;
    (which ? s2 : s1)[b * 64 + c] = v;
}

// ---------------------------------------------------------------------------
// Kernel 2: modulated + demodulated weights, bf16, layout [n][t*64 + i]
//   bid <  512 : conv1 (b = bid/64, o = bid%64)
//   512..767   : conv2 (b, o%32)
//   768..799   : shortcut weights (no modulation, just re-layout + bf16)
// ---------------------------------------------------------------------------
__global__ void style_w_kernel(const float* __restrict__ conv1_w,
                               const float* __restrict__ conv2_w,
                               const float* __restrict__ sc_w,
                               const float* __restrict__ s1,
                               const float* __restrict__ s2,
                               unsigned short* __restrict__ w1m,
                               unsigned short* __restrict__ w2m,
                               unsigned short* __restrict__ wsc) {
    __shared__ float red[256];
    int bid = blockIdx.x, tid = threadIdx.x;
    if (bid < 768) {
        const float* W; const float* S; unsigned short* out;
        if (bid < 512) {
            int b = bid >> 6, o = bid & 63;
            W = conv1_w + o * 576; S = s1 + b * 64; out = w1m + (b * 64 + o) * 576;
        } else {
            int r = bid - 512; int b = r >> 5, o = r & 31;
            W = conv2_w + o * 576; S = s2 + b * 64; out = w2m + (b * 32 + o) * 576;
        }
        float ss = 0.f;
        for (int k = tid; k < 576; k += 256) {
            float v = W[k] * S[k / 9];          // k = i*9 + t
            ss += v * v;
        }
        red[tid] = ss; __syncthreads();
        for (int s = 128; s > 0; s >>= 1) {
            if (tid < s) red[tid] += red[tid + s];
            __syncthreads();
        }
        float d = rsqrtf(red[0] + 1e-8f);
        for (int k = tid; k < 576; k += 256) {  // k = t*64 + i (GEMM layout)
            int t = k >> 6, i = k & 63;
            out[k] = f2bf(W[i * 9 + t] * S[i] * d);
        }
    } else {
        int o = bid - 768;
        for (int k = tid; k < 576; k += 256) {
            int t = k >> 6, i = k & 63;
            wsc[o * 576 + k] = f2bf(sc_w[o * 576 + i * 9 + t]);
        }
    }
}

// ---------------------------------------------------------------------------
// Kernel 3: bilinear x2 upsample, align_corners=True, NCHW f32 -> NHWC bf16
// ---------------------------------------------------------------------------
__global__ void upsample_kernel(const float* __restrict__ in,
                                unsigned short* __restrict__ ups) {
    int yo = blockIdx.x;              // 0..255
    int b  = blockIdx.y;              // 0..7
    int c  = threadIdx.x & 63;
    int xq = threadIdx.x >> 6;
    float sy = yo * (127.0f / 255.0f);
    int y0 = (int)sy; if (y0 > 127) y0 = 127;
    int y1 = y0 + 1;  if (y1 > 127) y1 = 127;
    float wy = sy - (float)y0;
    const float* p0 = in + ((size_t)(b * 64 + c) * 128 + y0) * 128;
    const float* p1 = in + ((size_t)(b * 64 + c) * 128 + y1) * 128;
    unsigned short* orow = ups + ((size_t)(b * 256 + yo) * 256) * 64 + c;
    for (int x = xq; x < 256; x += 4) {
        float sx = x * (127.0f / 255.0f);
        int x0 = (int)sx; if (x0 > 127) x0 = 127;
        int x1 = x0 + 1;  if (x1 > 127) x1 = 127;
        float wx = sx - (float)x0;
        float r0 = p0[x0] * (1.f - wx) + p0[x1] * wx;
        float r1 = p1[x0] * (1.f - wx) + p1[x1] * wx;
        orow[(size_t)x * 64] = f2bf(r0 * (1.f - wy) + r1 * wy);
    }
}

// ---------------------------------------------------------------------------
// Shared conv machinery: LDS halo tile + WMMA bf16 fragments
// Tile: [3 rows][130 cols][72 ch-padded] bf16  (col 0 == x0-1)
// ---------------------------------------------------------------------------
#define TCH 72

static __device__ __forceinline__ void load_tile(unsigned short tile[3][130][TCH],
                                                 const unsigned short* __restrict__ src,
                                                 int b, int y, int x0, int tid) {
    for (int idx = tid; idx < 3 * 130 * 16; idx += 256) {
        int dy = idx / (130 * 16);
        int r  = idx - dy * (130 * 16);
        int cc = r >> 4;
        int c4 = (r & 15) * 4;
        int ys = y + dy - 1;
        int xs = x0 + cc - 1;
        uint2 v = make_uint2(0u, 0u);
        if ((unsigned)ys < 256u && (unsigned)xs < 256u)
            v = *reinterpret_cast<const uint2*>(src + (((size_t)(b * 256 + ys) * 256 + xs) * 64 + c4));
        *reinterpret_cast<uint2*>(&tile[dy][cc][c4]) = v;
    }
}

// A fragment (16x32 bf16, MxK). Lane: M=lane%16 selects cc; h=lane/16.
// u32 words of the K-row: [4h..4h+3] and [8+4h..8+4h+3].
static __device__ __forceinline__ v16bf load_afrag(const unsigned short tile[3][130][TCH],
                                                   int dy, int cc, int ci0, int h) {
    const int* base = reinterpret_cast<const int*>(&tile[dy][cc][0]) + (ci0 >> 1);
    v4i lo = *reinterpret_cast<const v4i*>(base + 4 * h);
    v4i hi = *reinterpret_cast<const v4i*>(base + 8 + 4 * h);
    v8i f;
    f[0] = lo[0]; f[1] = lo[1]; f[2] = lo[2]; f[3] = lo[3];
    f[4] = hi[0]; f[5] = hi[1]; f[6] = hi[2]; f[7] = hi[3];
    return __builtin_bit_cast(v16bf, f);
}

// B fragment (32x16 bf16, KxN). Lane: N=lane%16 (folded into wrow); h=lane/16.
// u32 words: [4h..4h+3] and [8+4h..8+4h+3].
static __device__ __forceinline__ v16bf load_bfrag(const unsigned short* __restrict__ wrow, int h) {
    const int* base = reinterpret_cast<const int*>(wrow);
    v4i lo = *reinterpret_cast<const v4i*>(base + 4 * h);
    v4i hi = *reinterpret_cast<const v4i*>(base + 8 + 4 * h);
    v8i f;
    f[0] = lo[0]; f[1] = lo[1]; f[2] = lo[2]; f[3] = lo[3];
    f[4] = hi[0]; f[5] = hi[1]; f[6] = hi[2]; f[7] = hi[3];
    return __builtin_bit_cast(v16bf, f);
}

// ---------------------------------------------------------------------------
// Kernel 4: shortcut conv 64->32 (+bias) -> f32 NCHW scratch
// ---------------------------------------------------------------------------
__global__ void __launch_bounds__(256) scconv_kernel(const unsigned short* __restrict__ ups,
                                                     const unsigned short* __restrict__ wsc,
                                                     const float* __restrict__ sc_b,
                                                     float* __restrict__ scs) {
    __shared__ unsigned short tile[3][130][TCH];
    __shared__ float smout[128][33];
    int x0 = blockIdx.x * 128, y = blockIdx.y, b = blockIdx.z;
    int tid = threadIdx.x;
    load_tile(tile, ups, b, y, x0, tid);
    __syncthreads();

    int lane = tid & 31, wid = tid >> 5;
    int ln = lane & 15, h = lane >> 4;
    int n0 = (wid & 1) * 16;
    int px = (wid >> 1) * 32;
    int n = n0 + ln;
    v8f acc[2] = {};
    const unsigned short* wbase = wsc + (size_t)n * 576;
    for (int kb = 0; kb < 18; ++kb) {
        int t = kb >> 1, ci0 = (kb & 1) * 32;
        int dy = t / 3, dx = t % 3;
        v16bf bf = load_bfrag(wbase + t * 64 + ci0, h);
#pragma unroll
        for (int m = 0; m < 2; ++m) {
            v16bf af = load_afrag(tile, dy, px + m * 16 + ln + dx, ci0, h);
            acc[m] = __builtin_amdgcn_wmma_f32_16x16x32_bf16(false, af, false, bf,
                                                             (short)0, acc[m], false, false);
        }
    }
    float bias = sc_b[n];
#pragma unroll
    for (int m = 0; m < 2; ++m)
#pragma unroll
        for (int r = 0; r < 8; ++r)
            smout[px + m * 16 + r + 8 * h][n] = acc[m][r] + bias;
    __syncthreads();
    for (int it = 0; it < 16; ++it) {
        int nn = it * 2 + (tid >> 7);
        int p  = tid & 127;
        scs[((size_t)(b * 32 + nn) * 256 + y) * 256 + x0 + p] = smout[p][nn];
    }
}

// ---------------------------------------------------------------------------
// Kernel 5: deterministic BN reduction (per-channel mean & invstd, f64 accum)
// ---------------------------------------------------------------------------
__global__ void bn_reduce_kernel(const float* __restrict__ scs, float* __restrict__ stats) {
    __shared__ double rs[256], rq[256];
    int n = blockIdx.x, tid = threadIdx.x;
    double s = 0.0, q = 0.0;
    for (int b = 0; b < 8; ++b) {
        const float* p = scs + (size_t)(b * 32 + n) * 65536;
        for (int i = tid; i < 65536; i += 256) {
            double v = (double)p[i];
            s += v; q += v * v;
        }
    }
    rs[tid] = s; rq[tid] = q; __syncthreads();
    for (int st = 128; st > 0; st >>= 1) {
        if (tid < st) { rs[tid] += rs[tid + st]; rq[tid] += rq[tid + st]; }
        __syncthreads();
    }
    if (tid == 0) {
        double N = 8.0 * 65536.0;
        double mean = rs[0] / N;
        double var  = rq[0] / N - mean * mean;
        stats[n]      = (float)mean;
        stats[32 + n] = rsqrtf((float)var + 1e-5f);
    }
}

// ---------------------------------------------------------------------------
// Kernel 6: style conv1 64->64, mish -> NHWC bf16
// ---------------------------------------------------------------------------
__global__ void __launch_bounds__(256) conv1_kernel(const unsigned short* __restrict__ ups,
                                                    const unsigned short* __restrict__ w1m,
                                                    unsigned short* __restrict__ act1) {
    __shared__ unsigned short tile[3][130][TCH];
    int x0 = blockIdx.x * 128, y = blockIdx.y, b = blockIdx.z;
    int tid = threadIdx.x;
    load_tile(tile, ups, b, y, x0, tid);
    __syncthreads();

    int lane = tid & 31, wid = tid >> 5;
    int ln = lane & 15, h = lane >> 4;
    int n0 = (wid & 3) * 16;
    int px = (wid >> 2) * 64;
    int n = n0 + ln;
    v8f acc[4] = {};
    const unsigned short* wbase = w1m + (size_t)(b * 64 + n) * 576;
    for (int kb = 0; kb < 18; ++kb) {
        int t = kb >> 1, ci0 = (kb & 1) * 32;
        int dy = t / 3, dx = t % 3;
        v16bf bf = load_bfrag(wbase + t * 64 + ci0, h);
#pragma unroll
        for (int m = 0; m < 4; ++m) {
            v16bf af = load_afrag(tile, dy, px + m * 16 + ln + dx, ci0, h);
            acc[m] = __builtin_amdgcn_wmma_f32_16x16x32_bf16(false, af, false, bf,
                                                             (short)0, acc[m], false, false);
        }
    }
#pragma unroll
    for (int m = 0; m < 4; ++m)
#pragma unroll
        for (int r = 0; r < 8; ++r) {
            int p = px + m * 16 + r + 8 * h;
            act1[((size_t)(b * 256 + y) * 256 + x0 + p) * 64 + n] = f2bf(mishf(acc[m][r]));
        }
}

// ---------------------------------------------------------------------------
// Kernel 7: style conv2 64->32, mish, + BN-affine shortcut add -> d_out NCHW
// ---------------------------------------------------------------------------
__global__ void __launch_bounds__(256) conv2_kernel(const unsigned short* __restrict__ act1,
                                                    const unsigned short* __restrict__ w2m,
                                                    const float* __restrict__ scs,
                                                    const float* __restrict__ stats,
                                                    const float* __restrict__ gamma,
                                                    const float* __restrict__ beta,
                                                    float* __restrict__ out) {
    __shared__ unsigned short tile[3][130][TCH];
    __shared__ float smout[128][33];
    int x0 = blockIdx.x * 128, y = blockIdx.y, b = blockIdx.z;
    int tid = threadIdx.x;
    load_tile(tile, act1, b, y, x0, tid);
    __syncthreads();

    int lane = tid & 31, wid = tid >> 5;
    int ln = lane & 15, h = lane >> 4;
    int n0 = (wid & 1) * 16;
    int px = (wid >> 1) * 32;
    int n = n0 + ln;
    v8f acc[2] = {};
    const unsigned short* wbase = w2m + (size_t)(b * 32 + n) * 576;
    for (int kb = 0; kb < 18; ++kb) {
        int t = kb >> 1, ci0 = (kb & 1) * 32;
        int dy = t / 3, dx = t % 3;
        v16bf bf = load_bfrag(wbase + t * 64 + ci0, h);
#pragma unroll
        for (int m = 0; m < 2; ++m) {
            v16bf af = load_afrag(tile, dy, px + m * 16 + ln + dx, ci0, h);
            acc[m] = __builtin_amdgcn_wmma_f32_16x16x32_bf16(false, af, false, bf,
                                                             (short)0, acc[m], false, false);
        }
    }
#pragma unroll
    for (int m = 0; m < 2; ++m)
#pragma unroll
        for (int r = 0; r < 8; ++r)
            smout[px + m * 16 + r + 8 * h][n] = mishf(acc[m][r]);
    __syncthreads();
    for (int it = 0; it < 16; ++it) {
        int nn = it * 2 + (tid >> 7);
        int p  = tid & 127;
        size_t gi = ((size_t)(b * 32 + nn) * 256 + y) * 256 + x0 + p;
        float scv = scs[gi];
        float v = smout[p][nn] + gamma[nn] * ((scv - stats[nn]) * stats[32 + nn]) + beta[nn];
        out[gi] = v;
    }
}

// ---------------------------------------------------------------------------
extern "C" void kernel_launch(void* const* d_in, const int* in_sizes, int n_in,
                              void* d_out, int out_size, void* d_ws, size_t ws_size,
                              hipStream_t stream) {
    const float* input   = (const float*)d_in[0];
    const float* z       = (const float*)d_in[1];
    const float* ws1_w   = (const float*)d_in[2];
    const float* ws1_b   = (const float*)d_in[3];
    const float* conv1_w = (const float*)d_in[4];
    const float* ws2_w   = (const float*)d_in[5];
    const float* ws2_b   = (const float*)d_in[6];
    const float* conv2_w = (const float*)d_in[7];
    const float* sc_w    = (const float*)d_in[8];
    const float* sc_b    = (const float*)d_in[9];
    const float* bn_g    = (const float*)d_in[10];
    const float* bn_b    = (const float*)d_in[11];
    float* out = (float*)d_out;

    char* ws = (char*)d_ws;
    float*          s1    = (float*)(ws + 0);                       //  2 KB
    float*          s2    = (float*)(ws + 2048);                    //  2 KB
    unsigned short* w1m   = (unsigned short*)(ws + 4096);           // 576 KB
    unsigned short* w2m   = (unsigned short*)(ws + 593920);         // 288 KB
    unsigned short* wsc   = (unsigned short*)(ws + 888832);         //  36 KB
    float*          stats = (float*)(ws + 925696);                  // 256 B
    unsigned short* ups   = (unsigned short*)(ws + (1u << 20));     //  64 MB
    unsigned short* act1  = (unsigned short*)(ws + 68157440ull);    //  64 MB
    float*          scs   = (float*)(ws + 135266304ull);            //  64 MB

    style_y_kernel<<<dim3(8, 2), 64, 0, stream>>>(z, ws1_w, ws1_b, ws2_w, ws2_b, s1, s2);
    style_w_kernel<<<800, 256, 0, stream>>>(conv1_w, conv2_w, sc_w, s1, s2, w1m, w2m, wsc);
    upsample_kernel<<<dim3(256, 8), 256, 0, stream>>>(input, ups);
    scconv_kernel<<<dim3(2, 256, 8), 256, 0, stream>>>(ups, wsc, sc_b, scs);
    bn_reduce_kernel<<<32, 256, 0, stream>>>(scs, stats);
    conv1_kernel<<<dim3(2, 256, 8), 256, 0, stream>>>(ups, w1m, act1);
    conv2_kernel<<<dim3(2, 256, 8), 256, 0, stream>>>(act1, w2m, scs, stats, bn_g, bn_b, out);
}
